// RNNModel_15187004358759
// MI455X (gfx1250) — matrix operand order repaired and checked by reading, not compiled, not run on hardware
//
#include <hip/hip_runtime.h>

// ---------------------------------------------------------------------------
// LSTM LM decode for MI455X (gfx1250, wave32, WMMA).
//   L=256 seq, B=16 batch, V=32000 vocab, H=512 hidden, G=4H=2048 gates.
// Phases:
//   1) embed_kernel : embB[l,b,:] = W_ih[:,tok] + b_ih + b_hh      (fp32, ws)
//   2) conv_kernel  : W_dec fp32 -> bf16                            (ws)
//   3) lstm_kernel  : persistent, 32 blocks x 4 waves, 256 steps,
//                     v_wmma_f32_16x16x32_bf16 per gate tile,
//                     global arrive/spin barrier per step, c in regs
//   4) decode_kernel: hs_bf16 @ Wdec_bf16^T + b_dec -> logits fp32,
//                     double-buffered GLOBAL_LOAD_ASYNC_TO_LDS_B128 staging,
//                     256x128 block tile, 16 wmma / wave / K-chunk,
//                     non-temporal logits stores
// ---------------------------------------------------------------------------

typedef __bf16 bf16_t;
typedef __attribute__((ext_vector_type(16))) __bf16 v16bf;
typedef __attribute__((ext_vector_type(8)))  __bf16 v8bf;
typedef __attribute__((ext_vector_type(8)))  float  v8f;
typedef __attribute__((ext_vector_type(4)))  int    v4i;

#define LL 256
#define BB 16
#define VV 32000
#define HH 512
#define GG 2048
#define LB (LL*BB)          // 4096
#define NBLK 32             // LSTM blocks (hidden slices of 16)

// ---- async global->LDS (gfx1250 GLOBAL_LOAD_ASYNC_TO_LDS_B128, ASYNCcnt) ---
#if __has_builtin(__builtin_amdgcn_global_load_async_to_lds_b128) && \
    __has_builtin(__builtin_amdgcn_s_wait_asynccnt)
#define HAVE_ASYNC_LDS 1
#else
#define HAVE_ASYNC_LDS 0
#endif

typedef __attribute__((address_space(1))) v4i* as1_v4i_p;
typedef __attribute__((address_space(3))) v4i* as3_v4i_p;

__device__ __forceinline__ void cp16_g2lds(void* dst_lds, const void* src_g) {
#if HAVE_ASYNC_LDS
    __builtin_amdgcn_global_load_async_to_lds_b128(
        (as1_v4i_p)(src_g), (as3_v4i_p)(dst_lds), 0, 0);
#else
    *(v8bf*)dst_lds = *(const v8bf*)src_g;
#endif
}

__device__ __forceinline__ void wait_cp_done() {
#if HAVE_ASYNC_LDS
    __builtin_amdgcn_s_wait_asynccnt(0);
#endif
}

__device__ __forceinline__ bf16_t f2bf(float f) {
    union { float f; unsigned u; } x; x.f = f;
    unsigned r = x.u + 0x7FFFu + ((x.u >> 16) & 1u);   // round-nearest-even
    union { unsigned short s; bf16_t b; } y; y.s = (unsigned short)(r >> 16);
    return y.b;
}

// A fragment (16x32 bf16, M x K) from an LDS/row-major buffer, stride in elems.
// Lane<16: M=lane, K=0..15; lane>=16: M=lane-16, K=16..31.
__device__ __forceinline__ v16bf load_A_frag(const bf16_t* base, int k0, int strideEl) {
    int lane = threadIdx.x & 31;
    int row  = lane & 15;
    int koff = (lane < 16) ? 0 : 16;
    const bf16_t* p = base + row * strideEl + k0 + koff;
    union { v16bf v; v8bf h[2]; } u;
    u.h[0] = *(const v8bf*)(p);
    u.h[1] = *(const v8bf*)(p + 8);
    return u.v;
}

// B fragment (32x16 bf16, K x N) from row-major [n][k] buffer.
// Lane<16: N=lane, K=k0..k0+15; lane>=16: N=lane-16, K=k0+16..k0+31.
__device__ __forceinline__ v16bf load_B_frag(const bf16_t* base, int k0, int strideEl) {
    int lane = threadIdx.x & 31;
    int n    = lane & 15;
    int koff = (lane < 16) ? 0 : 16;
    const bf16_t* p = base + n * strideEl + k0 + koff;
    union { v16bf v; v8bf h[2]; } u;
    u.h[0] = *(const v8bf*)(p);
    u.h[1] = *(const v8bf*)(p + 8);
    return u.v;
}

__device__ __forceinline__ v8f wmma_bf16(v16bf a, v16bf b, v8f c) {
    return __builtin_amdgcn_wmma_f32_16x16x32_bf16(false, a, false, b,
                                                   (short)0, c, false, false);
}

// ---------------------------------------------------------------------------
// 1) Embedding gather + fused bias: embB[lb*2048 + col]
__global__ void embed_kernel(const int* __restrict__ tokens,
                             const float* __restrict__ W_ih,
                             const float* __restrict__ b_ih,
                             const float* __restrict__ b_hh,
                             float* __restrict__ embB) {
    int lb  = blockIdx.x;              // l*B + b
    int tok = tokens[lb];
#pragma unroll
    for (int i = 0; i < 8; ++i) {
        int col = threadIdx.x + i * 256;
        embB[(size_t)lb * GG + col] =
            W_ih[(size_t)col * VV + tok] + b_ih[col] + b_hh[col];
    }
}

// 2) fp32 -> bf16 convert (W_dec)
__global__ void conv_kernel(const float* __restrict__ src,
                            bf16_t* __restrict__ dst) {
    size_t i0 = ((size_t)blockIdx.x * 256 + threadIdx.x) * 16;
#pragma unroll
    for (int i = 0; i < 16; ++i) dst[i0 + i] = f2bf(src[i0 + i]);
}

// ---------------------------------------------------------------------------
// 3) Persistent LSTM. 32 blocks x 128 threads. Block bn owns hidden cols
//    [bn*16, bn*16+16). Wave w computes gate w (i/f/g/o) for those cols.
__global__ void __launch_bounds__(128, 1)
lstm_kernel(const float* __restrict__ W_hh,
            const float* __restrict__ embB,
            bf16_t* __restrict__ hbuf,      // ping-pong 2 x [16][512] bf16
            bf16_t* __restrict__ hs,        // [4096][512] bf16 (decoder A)
            unsigned* __restrict__ bar,     // global barrier counter (zeroed)
            float* __restrict__ hL,         // [16][512] fp32 out
            float* __restrict__ cL) {       // [16][512] fp32 out
    __shared__ alignas(16) bf16_t sA[16 * 520];  // staged h, padded stride 520
    __shared__ float  sG[4 * 16 * 16];           // gate tiles i,f,g,o

    const int tid  = threadIdx.x;
    const int w    = tid >> 5;              // wave = gate index
    const int lane = tid & 31;
    const int n0   = blockIdx.x * 16;

    // Preload this wave's W_hh gate tile as bf16 B-fragments (K=512 -> 16).
    const int rowBase = w * HH + n0;        // W_hh row = gate column index
    v16bf Bf[16];
#pragma unroll
    for (int c = 0; c < 16; ++c) {
        int nn   = lane & 15;
        int koff = (lane < 16) ? 0 : 16;
        const float* p = W_hh + (size_t)(rowBase + nn) * HH + c * 32 + koff;
        v16bf t;
#pragma unroll
        for (int i = 0; i < 16; ++i) t[i] = f2bf(p[i]);
        Bf[c] = t;
    }

    // Two cells per thread; fp32 cell state in registers.
    const int cA = tid, cB = tid + 128;
    const int batA = cA >> 4, colA = cA & 15;
    const int batB = cB >> 4, colB = cB & 15;
    float csA = 0.f, csB = 0.f;

    for (int l = 0; l < LL; ++l) {
        const int par = l & 1;
        const bf16_t* hin  = hbuf + par * (BB * HH);
        bf16_t*       hout = hbuf + (par ^ 1) * (BB * HH);

        // Stage h (16x512 bf16) into LDS: 128 threads x 128 B.
        {
            int bat = tid >> 3, seg = tid & 7;
            const v8bf* src = (const v8bf*)(hin + bat * HH + seg * 64);
            v8bf*       dst = (v8bf*)(sA + bat * 520 + seg * 64);
#pragma unroll
            for (int i = 0; i < 8; ++i) dst[i] = src[i];
        }
        __syncthreads();

        // gates(w) = h @ W_hh[gate w]^T  (K = 512, 16 bf16 WMMA steps)
        v8f acc = {};
#pragma unroll
        for (int c = 0; c < 16; ++c) {
            v16bf a = load_A_frag(sA, c * 32, 520);
            acc = wmma_bf16(a, Bf[c], acc);
        }
        // Scatter accumulator to sG[gate][bat][col].
        {
            int colC = lane & 15, batO = (lane < 16) ? 0 : 8;
#pragma unroll
            for (int r = 0; r < 8; ++r)
                sG[w * 256 + (batO + r) * 16 + colC] = acc[r];
        }
        __syncthreads();

        // Cell update (fp32): gates += emb + bias (prefused in embB).
        const float* eb = embB + (size_t)l * BB * GG;
#pragma unroll
        for (int cell = 0; cell < 2; ++cell) {
            int bat = cell ? batB : batA;
            int col = cell ? colB : colA;
            float gi = sG[0 * 256 + bat * 16 + col] + eb[bat * GG + 0 * HH + n0 + col];
            float gf = sG[1 * 256 + bat * 16 + col] + eb[bat * GG + 1 * HH + n0 + col];
            float gg = sG[2 * 256 + bat * 16 + col] + eb[bat * GG + 2 * HH + n0 + col];
            float go = sG[3 * 256 + bat * 16 + col] + eb[bat * GG + 3 * HH + n0 + col];
            float i_ = 1.f / (1.f + __expf(-gi));
            float f_ = 1.f / (1.f + __expf(-gf));
            float g_ = tanhf(gg);
            float o_ = 1.f / (1.f + __expf(-go));
            float cs = f_ * (cell ? csB : csA) + i_ * g_;
            if (cell) csB = cs; else csA = cs;
            float h_ = o_ * tanhf(cs);
            bf16_t hb = f2bf(h_);
            hout[bat * HH + n0 + col] = hb;
            hs[((size_t)l * BB + bat) * HH + n0 + col] = hb;
            if (l == LL - 1) {
                hL[bat * HH + n0 + col] = h_;
                cL[bat * HH + n0 + col] = cs;
            }
        }

        // Device-wide step barrier: monotone counter, no reset race.
        __threadfence();
        __syncthreads();
        if (tid == 0) {
            __hip_atomic_fetch_add(bar, 1u, __ATOMIC_RELEASE, __HIP_MEMORY_SCOPE_AGENT);
            unsigned tgt = (unsigned)NBLK * (unsigned)(l + 1);
            while (__hip_atomic_load(bar, __ATOMIC_ACQUIRE, __HIP_MEMORY_SCOPE_AGENT) < tgt)
                __builtin_amdgcn_s_sleep(1);
        }
        __syncthreads();
    }
}

// ---------------------------------------------------------------------------
// 4) Decoder GEMM: logits[4096][32000] = hs @ Wdec^T + b_dec.
//    Block tile 256(M) x 128(N), 8 waves; wave w -> rows w*32..w*32+31,
//    two A-fragments share each B-fragment (16 wmma / chunk / wave).
//    Double-buffered LDS, async global->LDS staging when available.
#define DEC_AS (256 * 40)   // one A buffer, elements (padded stride 40)
#define DEC_BS (128 * 40)   // one B buffer, elements
__global__ void __launch_bounds__(256)
decode_kernel(const bf16_t* __restrict__ hs,
              const bf16_t* __restrict__ Wd,
              const float* __restrict__ bdec,
              float* __restrict__ out) {
    __shared__ alignas(16) bf16_t sA[2 * DEC_AS];   // 40 KB
    __shared__ alignas(16) bf16_t sB[2 * DEC_BS];   // 20 KB

    const int tid  = threadIdx.x;
    const int w    = tid >> 5;
    const int lane = tid & 31;
    const int mBase = blockIdx.y * 256;
    const int nBase = blockIdx.x * 128;
    const int m0    = mBase + w * 32;

    v8f acc0[8], acc1[8];
#pragma unroll
    for (int j = 0; j < 8; ++j) { acc0[j] = (v8f){}; acc1[j] = (v8f){}; }

    // Stage one K-chunk (32 wide): A 256x32, B 128x32, 16 B per copy.
    auto stage = [&](int kc, int buf) {
        const int k0 = kc * 32;
        bf16_t* dA = sA + buf * DEC_AS;
        bf16_t* dB = sB + buf * DEC_BS;
#pragma unroll
        for (int i = 0; i < 2; ++i) {       // A: 512 pieces / 256 threads
            int p = tid + i * 256;
            int row = p >> 1, half = p & 1;
            cp16_g2lds(dA + row * 40 + half * 16,
                       hs + (size_t)(mBase + row) * HH + k0 + half * 16);
        }
        {                                    // B: 256 pieces / 256 threads
            int row = tid >> 1, half = tid & 1;
            cp16_g2lds(dB + row * 40 + half * 16,
                       Wd + (size_t)(nBase + row) * HH + k0 + half * 16);
        }
    };

    stage(0, 0);
    for (int kc = 0; kc < 16; ++kc) {
        wait_cp_done();                      // this wave's copies landed
        __syncthreads();                     // all waves' data visible;
                                             // prior-buffer reads retired
        if (kc < 15) stage(kc + 1, (kc + 1) & 1);   // overlap with WMMAs

        const bf16_t* bA = sA + (kc & 1) * DEC_AS;
        const bf16_t* bB = sB + (kc & 1) * DEC_BS;
        v16bf a0 = load_A_frag(bA + (size_t)(w * 32) * 40, 0, 40);
        v16bf a1 = load_A_frag(bA + (size_t)(w * 32 + 16) * 40, 0, 40);
#pragma unroll
        for (int j = 0; j < 8; ++j) {
            v16bf b = load_B_frag(bB + (size_t)(j * 16) * 40, 0, 40);
            acc0[j] = wmma_bf16(a0, b, acc0[j]);
            acc1[j] = wmma_bf16(a1, b, acc1[j]);
        }
    }

    // Epilogue: C layout -> VGPR r: (m = r or r+8, n = lane%16).
    // Logits are written once and never re-read: non-temporal stores.
    const int colC = lane & 15, batO = (lane < 16) ? 0 : 8;
#pragma unroll
    for (int j = 0; j < 8; ++j) {
        int n = nBase + j * 16 + colC;
        float bb = bdec[n];
#pragma unroll
        for (int r = 0; r < 8; ++r) {
            __builtin_nontemporal_store(acc0[j][r] + bb,
                &out[(size_t)(m0 + batO + r) * VV + n]);
            __builtin_nontemporal_store(acc1[j][r] + bb,
                &out[(size_t)(m0 + 16 + batO + r) * VV + n]);
        }
    }
}

// ---------------------------------------------------------------------------
extern "C" void kernel_launch(void* const* d_in, const int* in_sizes, int n_in,
                              void* d_out, int out_size, void* d_ws, size_t ws_size,
                              hipStream_t stream) {
    (void)in_sizes; (void)n_in; (void)out_size; (void)ws_size;
    const int*   tokens = (const int*)  d_in[0];
    const float* W_ih   = (const float*)d_in[1];
    const float* W_hh   = (const float*)d_in[2];
    const float* b_ih   = (const float*)d_in[3];
    const float* b_hh   = (const float*)d_in[4];
    const float* W_dec  = (const float*)d_in[5];
    const float* b_dec  = (const float*)d_in[6];
    float* out = (float*)d_out;

    // Workspace layout (bytes):
    //   [0,512)            barrier counter
    //   [512,33280)        h ping-pong: 2 x 16x512 bf16
    //   [33280, +33.5MB)   embB fp32 [L][B][2048]
    //   then hs bf16 [4096][512], then Wdec bf16 [32000][512]. ~70.6 MB total.
    char* ws = (char*)d_ws;
    unsigned* bar  = (unsigned*)ws;
    bf16_t*   hbuf = (bf16_t*)(ws + 512);
    float*    embB = (float*) (ws + 33280);
    bf16_t*   hsb  = (bf16_t*)(ws + 33280 + (size_t)LB * GG * 4);
    bf16_t*   Wdb  = (bf16_t*)(ws + 33280 + (size_t)LB * GG * 4 + (size_t)LB * HH * 2);

    // Re-init barrier + h0=0 every call (ws is poisoned, never restored).
    (void)hipMemsetAsync(ws, 0, 33280, stream);

    embed_kernel<<<dim3(LB), dim3(256), 0, stream>>>(tokens, W_ih, b_ih, b_hh, embB);
    conv_kernel<<<dim3((VV * HH) / (256 * 16)), dim3(256), 0, stream>>>(W_dec, Wdb);

    float* hL = out + (size_t)LB * VV;          // [1][16][512]
    float* cL = hL + (size_t)BB * HH;           // [1][16][512]
    lstm_kernel<<<dim3(NBLK), dim3(128), 0, stream>>>(W_hh, embB, hbuf, hsb, bar, hL, cL);

    decode_kernel<<<dim3(VV / 128, LB / 256), dim3(256), 0, stream>>>(hsb, Wdb, b_dec, out);
}
